// TTM_23811298689124
// MI455X (gfx1250) — matrix operand order repaired and checked by reading, not compiled, not run on hardware
//
#include <hip/hip_runtime.h>
#include <hip/hip_bf16.h>

// ---------------------------------------------------------------------------
// TTM restructured for CDNA5 (gfx1250):
//  * q/k/softmax path is provably dead (softmax of a scalar == 1) -> skipped.
//  * timestep scan collapses to an exclusive prefix-sum over 30 rows per layer,
//    turning every matvec into a batch-30 (pad 32) GEMM mapped onto
//    v_wmma_f32_16x16x32_bf16 (M=32 -> two 16-row tiles).
//  * Weights are converted f32->bf16 ONCE per launch (~7.5us at 23.3TB/s) so
//    the GEMM inner loop is pure b128 loads + WMMA (no conversion VALU).
//    58MB of bf16 weights stay resident in the 192MB L2 across layers.
//  * Fallback template keeps in-register conversion if ws_size is too small.
// ---------------------------------------------------------------------------

typedef __attribute__((ext_vector_type(16))) __bf16 v16bf;
typedef __attribute__((ext_vector_type(8)))  float  v8f;

union AFrag { v16bf v; int4 i[2]; };
union BFrag { v16bf v; int4 i[2]; unsigned short u[16]; };

__device__ __forceinline__ unsigned short f2bf(float f) {
    union { float f; unsigned u; } c; c.f = f;
    unsigned r = c.u + 0x7FFFu + ((c.u >> 16) & 1u);   // round-to-nearest-even
    return (unsigned short)(r >> 16);
}

// Store one bf16 scalar (row gm in [0,32), column/K-index k) into the WMMA
// A-fragment layout used by this file (contiguous K, same order as B):
//   lane l<16  : row=l,   K = e                (K 0..15)
//   lane l>=16 : row=l-16,K = e + 16           (K 16..31)
// buffer index = ((kt*2 + mt)*32 + lane)*16 + e,  kt=k/32, mt=gm/16
__device__ __forceinline__ void frag_store(unsigned short* frag, int gm, int k,
                                           unsigned short val) {
    int kt = k >> 5, ko = k & 31;
    int mt = gm >> 4;
    int h  = ko >> 4;
    int e  = ko & 15;
    int l  = h * 16 + (gm & 15);
    frag[((((size_t)kt * 2 + mt) * 32) + l) * 16 + e] = val;
}

// ---------------------------------------------------------------------------
// Bulk f32 -> bf16 weight conversion (8 elements / thread, vectorized).
// ---------------------------------------------------------------------------
__global__ __launch_bounds__(256) void cvt_bf16_kernel(
    const float* __restrict__ in, unsigned short* __restrict__ out, int n8) {
    int i = blockIdx.x * 256 + threadIdx.x;
    if (i >= n8) return;
    const float4* p = (const float4*)in + (size_t)i * 2;
    float4 a = p[0], b = p[1];
    union { unsigned short u[8]; uint4 v; } o;
    o.u[0] = f2bf(a.x); o.u[1] = f2bf(a.y); o.u[2] = f2bf(a.z); o.u[3] = f2bf(a.w);
    o.u[4] = f2bf(b.x); o.u[5] = f2bf(b.y); o.u[6] = f2bf(b.z); o.u[7] = f2bf(b.w);
    ((uint4*)out)[i] = o.v;
}

// ---------------------------------------------------------------------------
// xcf: segment-average x (C=32,S=32,T=360) -> rows (30,1024), write A-fragment
// ---------------------------------------------------------------------------
__global__ __launch_bounds__(256) void xcf_frag_kernel(
    const float* __restrict__ x, unsigned short* __restrict__ frag) {
    int idx = blockIdx.x * 256 + threadIdx.x;        // 32*1024
    if (idx >= 32 * 1024) return;
    int r = idx >> 10, m = idx & 1023;
    unsigned short v = 0;
    if (r < 30) {
        int s = m >> 5, c = m & 31;
        const float* p = x + (size_t)c * (32 * 360) + (size_t)s * 360 + r * 12;
        float acc = 0.f;
        #pragma unroll
        for (int g = 0; g < 12; ++g) acc += p[g];
        v = f2bf(acc * (1.0f / 12.0f));
    }
    frag_store(frag, r, m, v);
}

// ---------------------------------------------------------------------------
// LayerNorm over 1024 elems of row blockIdx.x; optional +residual; write frag.
// ---------------------------------------------------------------------------
__global__ __launch_bounds__(256) void ln_frag_kernel(
    const float* __restrict__ S, const float* __restrict__ g,
    const float* __restrict__ b, unsigned short* __restrict__ frag,
    int addResidual) {
    __shared__ float red[256];
    __shared__ float stat[2];
    int r = blockIdx.x, t = threadIdx.x;
    if (r >= 30) {
        #pragma unroll
        for (int j = 0; j < 4; ++j) frag_store(frag, r, t + 256 * j, 0);
        return;
    }
    const float* row = S + (size_t)r * 1024;
    float lv[4]; float sum = 0.f;
    #pragma unroll
    for (int j = 0; j < 4; ++j) { lv[j] = row[t + 256 * j]; sum += lv[j]; }
    red[t] = sum; __syncthreads();
    for (int o = 128; o > 0; o >>= 1) { if (t < o) red[t] += red[t + o]; __syncthreads(); }
    if (t == 0) stat[0] = red[0] * (1.0f / 1024.0f);
    __syncthreads();
    float mu = stat[0];
    float vs = 0.f;
    #pragma unroll
    for (int j = 0; j < 4; ++j) { float d = lv[j] - mu; vs += d * d; }
    red[t] = vs; __syncthreads();
    for (int o = 128; o > 0; o >>= 1) { if (t < o) red[t] += red[t + o]; __syncthreads(); }
    if (t == 0) stat[1] = red[0] * (1.0f / 1024.0f);
    __syncthreads();
    float inv = rsqrtf(stat[1] + 1e-5f);
    #pragma unroll
    for (int j = 0; j < 4; ++j) {
        int k = t + 256 * j;
        float o = (lv[j] - mu) * inv * g[k] + b[k];
        if (addResidual) o += lv[j];
        frag_store(frag, r, k, f2bf(o));
    }
}

// ---------------------------------------------------------------------------
// Exclusive prefix over the 30 time rows of V (the only sequential coupling).
// ---------------------------------------------------------------------------
__global__ __launch_bounds__(256) void prefix_frag_kernel(
    const float* __restrict__ V, unsigned short* __restrict__ frag) {
    int c = blockIdx.x * 256 + threadIdx.x;
    if (c >= 1024) return;
    float run = 0.f;
    for (int i = 0; i < 30; ++i) {
        frag_store(frag, i, c, f2bf(run));
        run += V[(size_t)i * 1024 + c];
    }
    frag_store(frag, 30, c, 0);
    frag_store(frag, 31, c, 0);
}

// ---------------------------------------------------------------------------
// GEMM: out(32,N) = Afrag(32,K bf16, fragment layout) * W(N,K)^T
// 256 threads = 8 waves = {2 N-subtiles of 16} x {4 K-splits}, LDS-reduced.
// BF16W=true : W is pre-converted bf16, inner loop = 6x b128 loads + 2 WMMA.
// BF16W=false: W is f32, converted in registers (ws_size fallback).
// mode: 0=none, 1=+sinusoidal PE, 2=exact GELU.  Optional bias / C-add.
// ---------------------------------------------------------------------------
#define WMMA_BF16 __builtin_amdgcn_wmma_f32_16x16x32_bf16

template <bool BF16W>
__global__ __launch_bounds__(256) void gemm_frag_kernel(
    const unsigned short* __restrict__ Af, const void* __restrict__ Wp,
    const float* __restrict__ bias, const float* __restrict__ Cadd,
    float* __restrict__ outF, unsigned short* __restrict__ outFrag,
    int N, int K, int mode) {
    __shared__ float sred[8 * 2 * 256];     // 8 waves x 2 M-tiles x (8 vgpr x 32 lanes)
    const int lane  = threadIdx.x & 31;
    const int wave  = threadIdx.x >> 5;
    const int nSub  = wave & 1;             // which 16-column subtile
    const int ks    = wave >> 1;            // K-split 0..3
    const int col   = blockIdx.x * 32 + nSub * 16 + (lane & 15);
    const int kh    = lane >> 4;            // B: lanes 0-15 -> K 0-15, 16-31 -> K 16-31
    const int ktN   = (K >> 2) >> 5;        // kt steps per split
    const int kt0   = ks * ktN;
    v8f acc0 = {0.f,0.f,0.f,0.f,0.f,0.f,0.f,0.f};
    v8f acc1 = acc0;

    const float*          wbaseF = (const float*)Wp          + (size_t)col * K + kh * 16;
    const unsigned short* wbaseH = (const unsigned short*)Wp + (size_t)col * K + kh * 16;

    #pragma unroll 2
    for (int kk = 0; kk < ktN; ++kk) {
        int kt = kt0 + kk;
        const unsigned short* ap = Af + (((size_t)kt * 2) * 32 + lane) * 16;
        AFrag a0, a1;
        a0.i[0] = *(const int4*)(ap);
        a0.i[1] = *(const int4*)(ap + 8);
        a1.i[0] = *(const int4*)(ap + 512);          // M-tile 1 (rows 16..31)
        a1.i[1] = *(const int4*)(ap + 520);
        BFrag bf;
        if (BF16W) {
            const unsigned short* wp = wbaseH + kt * 32;
            bf.i[0] = *(const int4*)(wp);
            bf.i[1] = *(const int4*)(wp + 8);
            if (kk + 1 < ktN) __builtin_prefetch(wp + 32, 0, 1);
        } else {
            const float* wp = wbaseF + kt * 32;
            if (kk + 1 < ktN) __builtin_prefetch(wp + 32, 0, 1);
            float4 f0 = *(const float4*)(wp);
            float4 f1 = *(const float4*)(wp + 4);
            float4 f2 = *(const float4*)(wp + 8);
            float4 f3 = *(const float4*)(wp + 12);
            bf.u[0]=f2bf(f0.x);  bf.u[1]=f2bf(f0.y);  bf.u[2]=f2bf(f0.z);  bf.u[3]=f2bf(f0.w);
            bf.u[4]=f2bf(f1.x);  bf.u[5]=f2bf(f1.y);  bf.u[6]=f2bf(f1.z);  bf.u[7]=f2bf(f1.w);
            bf.u[8]=f2bf(f2.x);  bf.u[9]=f2bf(f2.y);  bf.u[10]=f2bf(f2.z); bf.u[11]=f2bf(f2.w);
            bf.u[12]=f2bf(f3.x); bf.u[13]=f2bf(f3.y); bf.u[14]=f2bf(f3.z); bf.u[15]=f2bf(f3.w);
        }
        acc0 = WMMA_BF16(false, a0.v, false, bf.v, (short)0, acc0, false, false);
        acc1 = WMMA_BF16(false, a1.v, false, bf.v, (short)0, acc1, false, false);
    }
    #pragma unroll
    for (int r = 0; r < 8; ++r) sred[(wave * 2 + 0) * 256 + r * 32 + lane] = acc0[r];
    #pragma unroll
    for (int r = 0; r < 8; ++r) sred[(wave * 2 + 1) * 256 + r * 32 + lane] = acc1[r];
    __syncthreads();

    if (wave < 2) {                          // one epilogue wave per N-subtile
        const int myN  = wave;
        const int col2 = blockIdx.x * 32 + myN * 16 + (lane & 15);
        float bv   = bias ? bias[col2] : 0.0f;
        float peco = 0.0f;
        if (mode == 1)   // ang(i,n) = i * 10000^(-4*(n/2)/1024)
            peco = __expf(-(float)((col2 >> 1) * 4) * (9.2103403719761836f / 1024.0f));
        #pragma unroll
        for (int t = 0; t < 2; ++t) {
            #pragma unroll
            for (int r = 0; r < 8; ++r) {
                float xv = bv;
                #pragma unroll
                for (int s = 0; s < 4; ++s)
                    xv += sred[((s * 2 + myN) * 2 + t) * 256 + r * 32 + lane];
                int gm = t * 16 + (lane >> 4) * 8 + r;   // C/D layout: m = 8*(lane>=16)+vgpr
                if (Cadd) xv += Cadd[(size_t)gm * N + col2];
                if (mode == 2) xv = 0.5f * xv * (1.0f + erff(xv * 0.70710678118654752f));
                if (mode == 1) {
                    float ang = (float)gm * peco;
                    xv += (col2 & 1) ? __cosf(ang) : __sinf(ang);
                }
                if (outF)    outF[(size_t)gm * N + col2] = xv;
                if (outFrag) frag_store(outFrag, gm, col2, f2bf(xv));
            }
        }
    }
}

__global__ __launch_bounds__(256) void copy_out_kernel(
    const float* __restrict__ S, float* __restrict__ out) {
    int i = blockIdx.x * 256 + threadIdx.x;
    if (i < 30 * 1024) out[i] = S[i];
}

// ---------------------------------------------------------------------------
extern "C" void kernel_launch(void* const* d_in, const int* in_sizes, int n_in,
                              void* d_out, int out_size, void* d_ws, size_t ws_size,
                              hipStream_t stream) {
    (void)in_sizes; (void)n_in; (void)out_size;
    const float* x      = (const float*)d_in[0];
    const float* weight = (const float*)d_in[1];
    // d_in[2] (Wq), d_in[3] (Wk): dead code -- softmax of a scalar is 1.
    const float* Wv     = (const float*)d_in[4];
    const float* Wo     = (const float*)d_in[5];
    const float* ln_g   = (const float*)d_in[6];
    const float* ln_b   = (const float*)d_in[7];
    const float* lnz_g  = (const float*)d_in[8];
    const float* lnz_b  = (const float*)d_in[9];
    const float* fc1_w  = (const float*)d_in[10];
    const float* fc1_b  = (const float*)d_in[11];
    const float* fc2_w  = (const float*)d_in[12];
    const float* fc2_b  = (const float*)d_in[13];

    // ---- workspace carve ----
    float* S = (float*)d_ws;                         // (32,1024) f32 state
    float* V = S + 32 * 1024;                        // (32,1024) f32 per-layer v
    unsigned short* zfrag = (unsigned short*)(V + 32 * 1024);  // A-frag K=1024
    unsigned short* cfrag = zfrag + 32 * 1024;                 // A-frag K=1024
    unsigned short* sfrag = cfrag + 32 * 1024;                 // A-frag K=1024
    unsigned short* hfrag = sfrag + 32 * 1024;                 // A-frag K=4096
    // bf16 weight pool (only used when ws_size permits)
    unsigned short* w16_weight = hfrag + 32 * 4096;            // 1M
    unsigned short* w16_Wv  = w16_weight + (size_t)1024 * 1024;    // 10M
    unsigned short* w16_Wo  = w16_Wv + (size_t)10 * 1024 * 1024;   // 10M
    unsigned short* w16_fc1 = w16_Wo + (size_t)10 * 1024 * 1024;   // 4M
    unsigned short* w16_fc2 = w16_fc1 + (size_t)4096 * 1024;       // 4M
    const size_t actBytes = (size_t)(32 * 1024 * 2) * 4 +
                            (size_t)(32 * 1024 * 3 + 32 * 4096) * 2;
    const size_t wElems   = (size_t)1024 * 1024 + 2 * (size_t)10 * 1024 * 1024 +
                            2 * (size_t)4096 * 1024;
    const bool bf16w = ws_size >= actBytes + wElems * 2;

    const void *Wt, *Wvp, *Wop, *F1p, *F2p;
    if (bf16w) {
        cvt_bf16_kernel<<< 512, 256, 0, stream>>>(weight, w16_weight, (1024 * 1024) / 8);
        cvt_bf16_kernel<<<5120, 256, 0, stream>>>(Wv,     w16_Wv,  (10 * 1024 * 1024) / 8);
        cvt_bf16_kernel<<<5120, 256, 0, stream>>>(Wo,     w16_Wo,  (10 * 1024 * 1024) / 8);
        cvt_bf16_kernel<<<2048, 256, 0, stream>>>(fc1_w,  w16_fc1, (4096 * 1024) / 8);
        cvt_bf16_kernel<<<2048, 256, 0, stream>>>(fc2_w,  w16_fc2, (4096 * 1024) / 8);
        Wt = w16_weight; Wvp = w16_Wv; Wop = w16_Wo; F1p = w16_fc1; F2p = w16_fc2;
    } else {
        Wt = weight; Wvp = Wv; Wop = Wo; F1p = fc1_w; F2p = fc2_w;
    }

    #define GEMM(grid, A, Wq, bias_, cadd_, of_, ofr_, N_, K_, mode_, off_)          \
        do {                                                                          \
            if (bf16w)                                                                \
                gemm_frag_kernel<true><<<grid, 256, 0, stream>>>(                     \
                    A, (const void*)((const unsigned short*)Wq + (off_)), bias_,      \
                    cadd_, of_, ofr_, N_, K_, mode_);                                 \
            else                                                                      \
                gemm_frag_kernel<false><<<grid, 256, 0, stream>>>(                    \
                    A, (const void*)((const float*)Wq + (off_)), bias_,               \
                    cadd_, of_, ofr_, N_, K_, mode_);                                 \
        } while (0)

    // s0 = xcf @ weight^T + PE   (PE computed analytically in the epilogue)
    xcf_frag_kernel<<<128, 256, 0, stream>>>(x, zfrag);
    GEMM(32, zfrag, Wt, nullptr, nullptr, S, nullptr, 1024, 1024, /*PE*/1, (size_t)0);

    for (int a = 0; a < 10; ++a) {
        const size_t off = (size_t)a * 1024 * 1024;
        // z = LN(s)
        ln_frag_kernel<<<32, 256, 0, stream>>>(S, ln_g, ln_b, zfrag, 0);
        // v = z @ Wv^T   (all 30 timesteps batched)
        GEMM(32, zfrag, Wvp, nullptr, nullptr, V, nullptr, 1024, 1024, 0, off);
        // vcum_i = sum_{j<i} v_j  (the only sequential coupling)
        prefix_frag_kernel<<<4, 256, 0, stream>>>(V, cfrag);
        // s += vcum @ Wo^T
        GEMM(32, cfrag, Wop, nullptr, S, S, nullptr, 1024, 1024, 0, off);
        // t = LN(s) + s
        ln_frag_kernel<<<32, 256, 0, stream>>>(S, lnz_g, lnz_b, sfrag, 1);
        // h = gelu(t @ fc1^T + b1)   -> fragment for next GEMM
        GEMM(128, sfrag, F1p, fc1_b, nullptr, nullptr, hfrag, 4096, 1024, /*GELU*/2, (size_t)0);
        // s = h @ fc2^T + b2
        GEMM(32, hfrag, F2p, fc2_b, nullptr, S, nullptr, 1024, 4096, 0, (size_t)0);
    }
    copy_out_kernel<<<120, 256, 0, stream>>>(S, (float*)d_out);
    #undef GEMM
}
